// ScaledDotProductAttention_74947179315469
// MI455X (gfx1250) — matrix operand order repaired and checked by reading, not compiled, not run on hardware
//
#include <hip/hip_runtime.h>

// ---------------------------------------------------------------------------
// Types
// ---------------------------------------------------------------------------
typedef __bf16 v16bf __attribute__((ext_vector_type(16)));
typedef __bf16 v8bf  __attribute__((ext_vector_type(8), may_alias));
typedef float  v8f   __attribute__((ext_vector_type(8)));
typedef float  v4f   __attribute__((ext_vector_type(4), may_alias));
typedef unsigned int v8u __attribute__((ext_vector_type(8)));
typedef unsigned int v4u __attribute__((ext_vector_type(4)));
typedef int v8i __attribute__((ext_vector_type(8)));
typedef int v4i __attribute__((ext_vector_type(4)));

#define B_  4
#define S_  4096
#define D_  128
#define NT  (B_ * S_)

// LDS pitches (padded to avoid bank conflicts; TDM pad feature reproduces them)
#define KP  136   // 128 + 8 bf16 (pad 4 dwords per 64-dword row)
#define VP  40    // 32 + 8 bf16  (pad 4 dwords per 16-dword row)
#define WP  136

#if __has_builtin(__builtin_amdgcn_tensor_load_to_lds)
#define HAVE_TDM 1
#endif

// ---------------------------------------------------------------------------
// Helpers
// ---------------------------------------------------------------------------
static __device__ __forceinline__ v16bf join16(v8bf lo, v8bf hi) {
  v16bf r;
#pragma unroll
  for (int i = 0; i < 8; ++i) { r[i] = lo[i]; r[i + 8] = hi[i]; }
  return r;
}
static __device__ __forceinline__ v8bf ld8(const __bf16* p) { return *(const v8bf*)p; }
static __device__ __forceinline__ v8bf cvt8(const float* p) {
  v4f a = *(const v4f*)p, b = *(const v4f*)(p + 4);
  v8bf r;
#pragma unroll
  for (int i = 0; i < 4; ++i) { r[i] = (__bf16)a[i]; r[i + 4] = (__bf16)b[i]; }
  return r;
}
static __device__ __forceinline__ v8f wmma_bf16(v16bf a, v16bf b, v8f c) {
  return __builtin_amdgcn_wmma_f32_16x16x32_bf16(false, a, false, b, (short)0, c,
                                                 false, false);
}
static __device__ __forceinline__ float fast_exp2(float x) {
#if __has_builtin(__builtin_amdgcn_exp2f)
  return __builtin_amdgcn_exp2f(x);
#else
  return exp2f(x);
#endif
}
// ds_swizzle xor-mask patterns (and=0x1f, or=0): imm = (xor<<10)|0x1f
#define SWZF(x, imm) \
  __builtin_bit_cast(float, __builtin_amdgcn_ds_swizzle(__builtin_bit_cast(int, (x)), (imm)))

// ---------------------------------------------------------------------------
// TDM: 2D tile load (bf16 elements) with LDS row padding
//   pad_int_code: row length = 2^(code+1) dwords ; pad_amt_code: (code+1) dwords
//   6-arg toolchain form: (g0 u32x4, g1 i32x8, g2 i32x4, g3 i32x4, i32x8, cpol)
// ---------------------------------------------------------------------------
static __device__ __forceinline__ void tdm_load_2d(unsigned lds_off, const void* gptr,
                                                   unsigned tensor_d0, unsigned tensor_d1,
                                                   unsigned tile_d0, unsigned tile_d1,
                                                   unsigned stride0,
                                                   unsigned pad_int_code,
                                                   unsigned pad_amt_code) {
#ifdef HAVE_TDM
  unsigned long long ga = (unsigned long long)gptr;
  v4u g0;
  g0[0] = 1u;                                                    // count=1, user mode
  g0[1] = lds_off;                                               // LDS byte address
  g0[2] = (unsigned)ga;                                          // global addr [31:0]
  g0[3] = (unsigned)((ga >> 32) & 0x01ffffffu) | 0x80000000u;    // addr[56:32] | type=2
  v8i g1;
  g1[0] = (int)((1u << 16) |                                     // data_size = 2 bytes
                (1u << 20) |                                     // pad_enable
                (pad_int_code << 22) | (pad_amt_code << 25));
  g1[1] = (int)((tensor_d0 & 0xffffu) << 16);                    // tensor_dim0[15:0]
  g1[2] = (int)((tensor_d0 >> 16) | ((tensor_d1 & 0xffffu) << 16));
  g1[3] = (int)((tensor_d1 >> 16) | (tile_d0 << 16));            // tile_dim0
  g1[4] = (int)tile_d1;                                          // tile_dim1 (tile_dim2=0)
  g1[5] = (int)stride0;                                          // tensor_dim0_stride lo
  g1[6] = 0;
  g1[7] = 0;
  v4i z4 = {0, 0, 0, 0};
  v8i z8 = {0, 0, 0, 0, 0, 0, 0, 0};
  __builtin_amdgcn_tensor_load_to_lds(g0, g1, z4, z4, z8, 0);
#else
  (void)lds_off; (void)gptr; (void)tensor_d0; (void)tensor_d1;
  (void)tile_d0; (void)tile_d1; (void)stride0; (void)pad_int_code; (void)pad_amt_code;
#endif
}

// ---------------------------------------------------------------------------
// Kernel 1: QKV projections (bf16 WMMA, f32 accumulate)
//   Q stored pre-scaled by log2(e)/sqrt(H); V stored transposed [B, D, S]
// ---------------------------------------------------------------------------
__global__ __launch_bounds__(128)
void qkv_proj_kernel(const float* __restrict__ x,
                     const float* __restrict__ Wq, const float* __restrict__ bq,
                     const float* __restrict__ Wk, const float* __restrict__ bk,
                     const float* __restrict__ Wv, const float* __restrict__ bv,
                     __bf16* __restrict__ Qm, __bf16* __restrict__ Km,
                     __bf16* __restrict__ Vt) {
  __shared__ __align__(16) __bf16 sWt[128 * WP];
  const int tid = threadIdx.x;
  const int wave = tid >> 5, lane = tid & 31;
  const int laneN = lane & 15, g = lane >> 4;
  const int tok0 = blockIdx.x * 64;
  const int b = tok0 >> 12;
  const int sIn = (tok0 & (S_ - 1)) + wave * 16;   // s of tile base within batch
  const int trow = tok0 + wave * 16;               // flat token base of this wave's tile

  // A operand: x tile (16 tokens x 128) as bf16, 4 K-chunks of 32
  const float* xr = x + (size_t)(trow + laneN) * D_;
  v16bf xa[4];
#pragma unroll
  for (int c = 0; c < 4; ++c)
    xa[c] = join16(cvt8(xr + 32 * c + 16 * g), cvt8(xr + 32 * c + 16 * g + 8));

  const float* Ws[3] = {Wq, Wk, Wv};
  const float* bs[3] = {bq, bk, bv};
  const float kQscale = 0.08838834764831845f * 1.4426950408889634f; // log2e/sqrt(128)

  for (int m3 = 0; m3 < 3; ++m3) {
    const float* W = Ws[m3];
    // cooperative transpose into LDS: sWt[h][k] = W[k][h], h = tid
#pragma unroll 4
    for (int k = 0; k < 128; ++k)
      sWt[tid * WP + k] = (__bf16)W[k * 128 + tid];
    __syncthreads();

#pragma unroll
    for (int ht = 0; ht < 8; ++ht) {
      const int h = 16 * ht + laneN;
      v8f c = {};
#pragma unroll
      for (int cc = 0; cc < 4; ++cc) {
        // B operand: b[e] = W[32cc + 16g + e][h] (contiguous in sWt row h)
        v16bf bop = join16(ld8(&sWt[h * WP + 32 * cc + 16 * g]),
                           ld8(&sWt[h * WP + 32 * cc + 16 * g + 8]));
        c = wmma_bf16(xa[cc], bop, c);
      }
      const float bias = bs[m3][h];
      if (m3 == 0) {
        __bf16* op = Qm + (size_t)trow * D_ + h;
#pragma unroll
        for (int r = 0; r < 8; ++r)
          op[(size_t)(r + 8 * g) * D_] = (__bf16)((c[r] + bias) * kQscale);
      } else if (m3 == 1) {
        __bf16* op = Km + (size_t)trow * D_ + h;
#pragma unroll
        for (int r = 0; r < 8; ++r)
          op[(size_t)(r + 8 * g) * D_] = (__bf16)(c[r] + bias);
      } else {
        v8bf pk;
#pragma unroll
        for (int r = 0; r < 8; ++r) pk[r] = (__bf16)(c[r] + bias);
        *(v8bf*)(Vt + ((size_t)(b * D_ + h)) * S_ + sIn + 8 * g) = pk;
      }
    }
    __syncthreads();
  }
}

// ---------------------------------------------------------------------------
// Kernel 2: flash attention (transposed-score formulation)
//   St = K * Q^T via WMMA ; online softmax ; ctx^T = Vt * P^T via WMMA
// ---------------------------------------------------------------------------
#ifndef HAVE_TDM
static __device__ __forceinline__ void fb_copy(__bf16* dK, __bf16* dV,
                                               const __bf16* gK, const __bf16* gVcol,
                                               int tid) {
  // K: 32 rows x 128, dst pitch KP ; thread t -> row t>>2, quarter t&3
  {
    const int n = tid >> 2, q = tid & 3;
    const v8bf* s = (const v8bf*)(gK + n * 128 + q * 32);
    v8bf* d = (v8bf*)(dK + n * KP + q * 32);
#pragma unroll
    for (int i = 0; i < 4; ++i) d[i] = s[i];
  }
  // V: 128 rows x 32, src pitch S_, dst pitch VP ; thread t -> row t
  {
    const v8bf* s = (const v8bf*)(gVcol + (size_t)tid * S_);
    v8bf* d = (v8bf*)(dV + tid * VP);
#pragma unroll
    for (int i = 0; i < 4; ++i) d[i] = s[i];
  }
}
#endif

__global__ __launch_bounds__(128)
void flash_attn_kernel(const __bf16* __restrict__ Qm, const __bf16* __restrict__ Km,
                       const __bf16* __restrict__ Vt, float* __restrict__ partials) {
  __shared__ __align__(16) __bf16 sK[2][32 * KP];
  __shared__ __align__(16) __bf16 sV[2][128 * VP];

  const int tid = threadIdx.x;
  const int wave = tid >> 5, lane = tid & 31;
  const int laneN = lane & 15, g = lane >> 4;
  const int b = blockIdx.x >> 6;
  const int qblk = blockIdx.x & 63;
  const int q0 = qblk * 64 + wave * 16;

  // Q as WMMA B-operand: b[e] = Q[m][32c + 8g + e] (e<8), Q[m][32c + 16 + 8g + e-8] (e>=8), m = laneN
  const __bf16* qrow = Qm + (size_t)(b * S_ + q0 + laneN) * D_;
  v16bf qB[4];
#pragma unroll
  for (int c = 0; c < 4; ++c)
    qB[c] = join16(ld8(qrow + 32 * c + 8 * g), ld8(qrow + 32 * c + 16 + 8 * g));

  v8f acc[8];
#pragma unroll
  for (int i = 0; i < 8; ++i) acc[i] = (v8f){};
  float mi = -3.0e38f, li = 0.0f;

  const __bf16* Kbase = Km + (size_t)b * S_ * D_;
  const __bf16* Vbase = Vt + (size_t)b * D_ * S_;
  const unsigned skOff[2] = {(unsigned)(size_t)&sK[0][0], (unsigned)(size_t)&sK[1][0]};
  const unsigned svOff[2] = {(unsigned)(size_t)&sV[0][0], (unsigned)(size_t)&sV[1][0]};

  // ---- prologue: chunk 0 into buffer 0 ----
#ifdef HAVE_TDM
  if (wave == 0) {
    tdm_load_2d(skOff[0], Kbase, 128, S_, 128, 32, 128, 5, 3);   // K rows: 64dw + 4dw pad
    tdm_load_2d(svOff[0], Vbase, S_, 128, 32, 128, S_, 3, 3);    // V rows: 16dw + 4dw pad
    __builtin_amdgcn_s_wait_tensorcnt(0);
  }
#else
  fb_copy(&sK[0][0], &sV[0][0], Kbase, Vbase, tid);
#endif
  __syncthreads();

  const int NCH = S_ / 32;
#pragma unroll 1
  for (int kc = 0; kc < NCH; ++kc) {
    const int cur = kc & 1, nxt = cur ^ 1;
    const bool more = (kc + 1 < NCH);
#ifdef HAVE_TDM
    if (wave == 0 && more) {
      tdm_load_2d(skOff[nxt], Kbase + (size_t)(kc + 1) * 32 * D_, 128, S_, 128, 32, 128, 5, 3);
      tdm_load_2d(svOff[nxt], Vbase + (size_t)(kc + 1) * 32, S_, 128, 32, 128, S_, 3, 3);
    }
#endif
    // ---- St = K_chunk * Q^T : two 16x16 tiles, contraction over H=128 ----
    v8f st0 = {}, st1 = {};
    const __bf16* kl = &sK[cur][0];
#pragma unroll
    for (int cc = 0; cc < 4; ++cc) {
      v16bf a0 = join16(ld8(kl + laneN * KP + 32 * cc + 8 * g),
                        ld8(kl + laneN * KP + 32 * cc + 16 + 8 * g));
      st0 = wmma_bf16(a0, qB[cc], st0);
      v16bf a1 = join16(ld8(kl + (16 + laneN) * KP + 32 * cc + 8 * g),
                        ld8(kl + (16 + laneN) * KP + 32 * cc + 16 + 8 * g));
      st1 = wmma_bf16(a1, qB[cc], st1);
    }
    // lane holds St[n][m]: m=laneN, n = 16t + r + 8g

    // ---- online softmax (exp2 domain; scale folded into Q) ----
    float cmax = st0[0];
#pragma unroll
    for (int r = 1; r < 8; ++r) cmax = fmaxf(cmax, st0[r]);
#pragma unroll
    for (int r = 0; r < 8; ++r) cmax = fmaxf(cmax, st1[r]);
    cmax = fmaxf(cmax, SWZF(cmax, 0x401F));               // swap halves (xor 16)
    const float mnew = fmaxf(mi, cmax);
    const float alpha = fast_exp2(mi - mnew);
    float p0[8], p1[8], ls = 0.0f;
#pragma unroll
    for (int r = 0; r < 8; ++r) { p0[r] = fast_exp2(st0[r] - mnew); ls += p0[r]; }
#pragma unroll
    for (int r = 0; r < 8; ++r) { p1[r] = fast_exp2(st1[r] - mnew); ls += p1[r]; }
    ls += SWZF(ls, 0x401F);
    li = li * alpha + ls;
    mi = mnew;
#pragma unroll
    for (int i = 0; i < 8; ++i)
#pragma unroll
      for (int r = 0; r < 8; ++r) acc[i][r] *= alpha;

    // ---- build P^T B-operand: b[e] = P[m][e + 16g] via half-wave dword swap ----
    v16bf pv;
#pragma unroll
    for (int r = 0; r < 8; ++r) { pv[r] = (__bf16)p0[r]; pv[8 + r] = (__bf16)p1[r]; }
    v8u pk = __builtin_bit_cast(v8u, pv);
    unsigned rec0 = (unsigned)__builtin_amdgcn_ds_swizzle((int)(g ? pk[0] : pk[4]), 0x401F);
    unsigned rec1 = (unsigned)__builtin_amdgcn_ds_swizzle((int)(g ? pk[1] : pk[5]), 0x401F);
    unsigned rec2 = (unsigned)__builtin_amdgcn_ds_swizzle((int)(g ? pk[2] : pk[6]), 0x401F);
    unsigned rec3 = (unsigned)__builtin_amdgcn_ds_swizzle((int)(g ? pk[3] : pk[7]), 0x401F);
    v8u bd;
    bd[0] = g ? rec0 : pk[0]; bd[1] = g ? rec1 : pk[1];
    bd[2] = g ? rec2 : pk[2]; bd[3] = g ? rec3 : pk[3];
    bd[4] = g ? pk[4] : rec0; bd[5] = g ? pk[5] : rec1;
    bd[6] = g ? pk[6] : rec2; bd[7] = g ? pk[7] : rec3;
    v16bf pB = __builtin_bit_cast(v16bf, bd);

    // ---- ctx^T[d][m] += Vt_tile(16x32) * P^T(32x16) ----
    const __bf16* vl = &sV[cur][0];
#pragma unroll
    for (int dt = 0; dt < 8; ++dt) {
      v16bf aV = join16(ld8(vl + (16 * dt + laneN) * VP + 16 * g),
                        ld8(vl + (16 * dt + laneN) * VP + 16 * g + 8));
      acc[dt] = wmma_bf16(aV, pB, acc[dt]);
    }

#ifdef HAVE_TDM
    if (wave == 0 && more) __builtin_amdgcn_s_wait_tensorcnt(0);
#else
    if (more)
      fb_copy(&sK[nxt][0], &sV[nxt][0], Kbase + (size_t)(kc + 1) * 32 * D_,
              Vbase + (size_t)(kc + 1) * 32, tid);
#endif
    __syncthreads();
  }

  // ---- epilogue: out-partial[d] = sum_m ctx[d][m] / l[m] (deterministic) ----
  const float rl = 1.0f / li;
  float* prow = partials + (size_t)(blockIdx.x * 4 + wave) * D_;
#pragma unroll
  for (int dt = 0; dt < 8; ++dt) {
#pragma unroll
    for (int r = 0; r < 8; ++r) {
      float v = acc[dt][r] * rl;
      v += SWZF(v, 0x041F);  // xor 1
      v += SWZF(v, 0x081F);  // xor 2
      v += SWZF(v, 0x101F);  // xor 4
      v += SWZF(v, 0x201F);  // xor 8  -> sum over 16 queries of this half-group
      if (laneN == 0) prow[16 * dt + r + 8 * g] = v;
    }
  }
}

// ---------------------------------------------------------------------------
// Kernel 3: deterministic reduction of per-wave partials -> out[b][d] (mean)
// ---------------------------------------------------------------------------
__global__ __launch_bounds__(128)
void reduce_kernel(const float* __restrict__ partials, float* __restrict__ out) {
  const int b = blockIdx.x;     // 0..3
  const int d = threadIdx.x;    // 0..127
  const float* p = partials + (size_t)b * 256 * D_ + d;
  float s = 0.0f;
#pragma unroll 4
  for (int i = 0; i < 256; ++i) s += p[(size_t)i * D_];
  out[b * D_ + d] = s * (1.0f / (float)S_);
}

// ---------------------------------------------------------------------------
// Launch
// ---------------------------------------------------------------------------
extern "C" void kernel_launch(void* const* d_in, const int* in_sizes, int n_in,
                              void* d_out, int out_size, void* d_ws, size_t ws_size,
                              hipStream_t stream) {
  (void)in_sizes; (void)n_in; (void)out_size; (void)ws_size;
  const float* x  = (const float*)d_in[0];
  const float* Wq = (const float*)d_in[1];
  const float* bq = (const float*)d_in[2];
  const float* Wk = (const float*)d_in[3];
  const float* bk = (const float*)d_in[4];
  const float* Wv = (const float*)d_in[5];
  const float* bv = (const float*)d_in[6];
  float* out = (float*)d_out;

  char* ws = (char*)d_ws;
  __bf16* Qm = (__bf16*)(ws);
  __bf16* Km = (__bf16*)(ws + (size_t)NT * D_ * 2);
  __bf16* Vt = (__bf16*)(ws + (size_t)NT * D_ * 4);
  float* partials = (float*)(ws + (size_t)NT * D_ * 6);

  qkv_proj_kernel<<<dim3(NT / 64), dim3(128), 0, stream>>>(x, Wq, bq, Wk, bk, Wv, bv,
                                                           Qm, Km, Vt);
  flash_attn_kernel<<<dim3(B_ * (S_ / 64)), dim3(128), 0, stream>>>(Qm, Km, Vt, partials);
  reduce_kernel<<<dim3(B_), dim3(128), 0, stream>>>(partials, out);
}
